// BertLayer_50946902065725
// MI455X (gfx1250) — matrix operand order repaired and checked by reading, not compiled, not run on hardware
//
#include <hip/hip_runtime.h>
#include <cstdint>
#include <cstddef>

// ---------------------------------------------------------------------------
// BERT layer forward on gfx1250 (MI455X): bf16 WMMA GEMMs + flash attention.
// - WMMA operand fragments fed by 128-bit LDS loads (tiles laid out to match
//   the ISA wave32 operand layouts).
// - A/K tile staging uses gfx1250 async global->LDS copies (ASYNCcnt).
// ---------------------------------------------------------------------------

typedef __bf16 bf16;
typedef __bf16 v16bf __attribute__((ext_vector_type(16)));
typedef float  v8f   __attribute__((ext_vector_type(8)));

#define D_MODEL 1024
#define N_HEADS 16
#define HEAD_DIM 64
#define D_FF 4096

union FragU {
  v16bf v;
  uint4 q[2];
};

__device__ __forceinline__ v8f wmma_bf16(v16bf a, v16bf b, v8f c) {
  // D = A(16x32 bf16) * B(32x16 bf16) + C(16x16 f32)
  return __builtin_amdgcn_wmma_f32_16x16x32_bf16(
      /*neg_a=*/false, a, /*neg_b=*/false, b,
      /*c_mod=*/(short)0, c, /*reuse_a=*/false, /*reuse_b=*/false);
}

// Async copy of 32 bytes (this lane) from global to LDS; ISA applies the
// instruction offset to BOTH the LDS and the global address.
__device__ __forceinline__ void async_copy32(const bf16* gsrc, void* ldst) {
  unsigned lds_off = (unsigned)(uintptr_t)ldst;
  asm volatile(
      "global_load_async_to_lds_b128 %0, %1, off\n\t"
      "global_load_async_to_lds_b128 %0, %1, off offset:16"
      :
      : "v"(lds_off), "v"(gsrc)
      : "memory");
}

__device__ __forceinline__ void async_wait0() {
  asm volatile("s_wait_asynccnt 0x0" ::: "memory");
}

// ---------------------------------------------------------------------------
// fp32 -> bf16 elementwise conversion
// ---------------------------------------------------------------------------
__global__ void cvt_f32_bf16(const float* __restrict__ in,
                             bf16* __restrict__ out, int n) {
  int i = blockIdx.x * 256 + threadIdx.x;
  if (i < n) out[i] = (bf16)in[i];
}

// ---------------------------------------------------------------------------
// Tiled bf16 GEMM: C[M,N] = A[M,K] @ B[K,N] + bias, optional GELU(erf).
// Block: 256 threads (8 waves). Block tile 128x64. Wave tile 32x32 (2x2 WMMA).
// A tile staged row-major [128][32] via async global->LDS; B tile staged
// TRANSPOSED [64][32] so both operand fragments are 2x b128 LDS loads.
// ---------------------------------------------------------------------------
__global__ __launch_bounds__(256) void gemm_bf16_wmma(
    const bf16* __restrict__ A, const bf16* __restrict__ B,
    const float* __restrict__ bias,
    float* __restrict__ Cf, bf16* __restrict__ Cb,
    int M, int N, int K, int fuse_gelu)
{
  __shared__ bf16 sA[128][40];   // [m][k], row stride 80B (16B aligned)
  __shared__ bf16 sBt[64][40];   // [n][k] transposed, row stride 80B

  const int tid  = threadIdx.x;
  const int lane = tid & 31;
  const int wave = tid >> 5;
  const int half = lane >> 4;
  const int l16  = lane & 15;

  const int ntiles  = N >> 6;
  const int bn      = blockIdx.x % ntiles;
  const int bm      = blockIdx.x / ntiles;
  const int rowBase = bm * 128;
  const int colBase = bn * 64;

  const int wm = (wave >> 1) * 32;  // 0,32,64,96
  const int wn = (wave & 1) * 32;   // 0,32

  v8f acc[2][2];
  for (int i = 0; i < 2; ++i)
    for (int j = 0; j < 2; ++j)
      for (int e = 0; e < 8; ++e) acc[i][j][e] = 0.0f;

  const int ar = tid >> 1, ac = (tid & 1) * 16;  // A stage: 16 bf16/thread
  const int br = tid >> 3, bc = (tid & 7) * 8;   // B stage:  8 bf16/thread

  for (int k0 = 0; k0 < K; k0 += 32) {
    // ---- stage tiles ----
    {
      // A: async DMA straight into LDS (no VGPR round trip)
      async_copy32(&A[(size_t)(rowBase + ar) * K + k0 + ac], &sA[ar][ac]);
      // B: coalesced 8-wide global read, transposed scatter into LDS
      const bf16* srcB = &B[(size_t)(k0 + br) * N + colBase + bc];
      uint4 braw = *reinterpret_cast<const uint4*>(srcB);
      const bf16* be = reinterpret_cast<const bf16*>(&braw);
      for (int i = 0; i < 8; ++i) sBt[bc + i][br] = be[i];
    }
    // prefetch next K tile while this one is consumed
    if (k0 + 32 < K) {
      __builtin_prefetch(&A[(size_t)(rowBase + ar) * K + k0 + 32 + ac], 0, 3);
      __builtin_prefetch(&B[(size_t)(k0 + 32 + br) * N + colBase + bc], 0, 3);
    }
    async_wait0();          // this wave's async LDS writes are visible
    __syncthreads();        // all waves' tiles staged

    // ---- fragments: 2x b128 LDS loads each ----
    FragU af[2], bfr[2];
    for (int i = 0; i < 2; ++i) {
      const int rm = wm + i * 16 + l16;
      af[i].q[0] = *reinterpret_cast<const uint4*>(&sA[rm][half * 8]);
      af[i].q[1] = *reinterpret_cast<const uint4*>(&sA[rm][16 + half * 8]);
    }
    for (int j = 0; j < 2; ++j) {
      const int cn = wn + j * 16 + l16;
      bfr[j].q[0] = *reinterpret_cast<const uint4*>(&sBt[cn][half * 8]);
      bfr[j].q[1] = *reinterpret_cast<const uint4*>(&sBt[cn][16 + half * 8]);
    }
    for (int i = 0; i < 2; ++i)
      for (int j = 0; j < 2; ++j)
        acc[i][j] = wmma_bf16(af[i].v, bfr[j].v, acc[i][j]);
    __syncthreads();
  }

  // Epilogue: C layout VGPR r -> row r + 8*half, cols on lanes.
  for (int i = 0; i < 2; ++i)
    for (int j = 0; j < 2; ++j)
      for (int r = 0; r < 8; ++r) {
        const int row = rowBase + wm + i * 16 + half * 8 + r;
        const int col = colBase + wn + j * 16 + l16;
        float v = acc[i][j][r] + bias[col];
        if (fuse_gelu)
          v = 0.5f * v * (1.0f + erff(v * 0.70710678118654752f));
        if (Cf) Cf[(size_t)row * N + col] = v;
        if (Cb) Cb[(size_t)row * N + col] = (bf16)v;
      }
}

// ---------------------------------------------------------------------------
// Flash attention: one block per (b, h, 128 query rows). 8 waves x 16 rows.
// sK [key][dim] staged via async copies (it is already the transposed form
// for the QK^T B operand); sV staged transposed [dim][key] for PV.
// ---------------------------------------------------------------------------
__global__ __launch_bounds__(256) void flash_attn_wmma(
    const bf16* __restrict__ Q, const bf16* __restrict__ K,
    const bf16* __restrict__ V, const float* __restrict__ mask,
    bf16* __restrict__ ctx, int S)
{
  __shared__ bf16 sK[64][72];       // [key][dim], row stride 144B
  __shared__ bf16 sVt[64][72];      // [dim][key] transposed
  __shared__ bf16 sP[8][16][72];    // per-wave P tile (C->A relayout)

  const int tid  = threadIdx.x;
  const int lane = tid & 31;
  const int wave = tid >> 5;
  const int half = lane >> 4;
  const int l16  = lane & 15;

  const int qblocks = S >> 7;       // S/128
  const int qb = blockIdx.x % qblocks;
  const int bh = blockIdx.x / qblocks;
  const int h  = bh % N_HEADS;
  const int b  = bh / N_HEADS;

  const int q0 = qb * 128 + wave * 16;  // this wave's first query row

  // Q fragments: two 16B chunks per lane per 32-K step -> global b128 loads.
  FragU qf[2];
  {
    const size_t qbase = (size_t)(b * S + q0 + l16) * D_MODEL + h * HEAD_DIM;
    for (int f = 0; f < 2; ++f) {
      qf[f].q[0] =
          *reinterpret_cast<const uint4*>(&Q[qbase + f * 32 + half * 8]);
      qf[f].q[1] =
          *reinterpret_cast<const uint4*>(&Q[qbase + f * 32 + 16 + half * 8]);
    }
  }

  float m_run[8], l_run[8];
  v8f o[4];
  for (int r = 0; r < 8; ++r) { m_run[r] = -3.0e38f; l_run[r] = 0.0f; }
  for (int nt = 0; nt < 4; ++nt)
    for (int e = 0; e < 8; ++e) o[nt][e] = 0.0f;

  const int lr = tid >> 2, lc = (tid & 3) * 16;  // K/V stage: 16 bf16/thread

  for (int kb = 0; kb < S; kb += 64) {
    // ---- stage K (async, row-major) and V (register transpose) ----
    {
      const size_t g = (size_t)(b * S + kb + lr) * D_MODEL + h * HEAD_DIM + lc;
      async_copy32(&K[g], &sK[lr][lc]);
      uint4 vraw[2];
      vraw[0] = *reinterpret_cast<const uint4*>(&V[g]);
      vraw[1] = *reinterpret_cast<const uint4*>(&V[g + 8]);
      const bf16* ve = reinterpret_cast<const bf16*>(vraw);
      for (int i = 0; i < 16; ++i) sVt[lc + i][lr] = ve[i];
    }
    async_wait0();
    __syncthreads();

    // ---- S_tile(16x64) = Q @ K_blk^T ----
    v8f st[4];
    for (int nt = 0; nt < 4; ++nt)
      for (int e = 0; e < 8; ++e) st[nt][e] = 0.0f;
    for (int f = 0; f < 2; ++f)
      for (int nt = 0; nt < 4; ++nt) {
        FragU bK;
        bK.q[0] = *reinterpret_cast<const uint4*>(
            &sK[nt * 16 + l16][f * 32 + half * 8]);
        bK.q[1] = *reinterpret_cast<const uint4*>(
            &sK[nt * 16 + l16][f * 32 + 16 + half * 8]);
        st[nt] = wmma_bf16(qf[f].v, bK.v, st[nt]);
      }

    // ---- scale (1/sqrt(64)) + additive mask ----
    for (int nt = 0; nt < 4; ++nt)
      for (int r = 0; r < 8; ++r) {
        const int colg = kb + nt * 16 + l16;
        const int qrow = q0 + half * 8 + r;
        const float mv = mask[(size_t)b * S * S + (size_t)qrow * S + colg];
        st[nt][r] = st[nt][r] * 0.125f + (1.0f - mv) * -10000.0f;
      }

    // ---- online softmax (row stats per (r, lane-half), 16-lane reduce) ----
    for (int r = 0; r < 8; ++r) {
      float mx = fmaxf(fmaxf(st[0][r], st[1][r]), fmaxf(st[2][r], st[3][r]));
      for (int off = 1; off < 16; off <<= 1)
        mx = fmaxf(mx, __shfl_xor(mx, off, 32));
      const float Mn    = fmaxf(m_run[r], mx);
      const float alpha = __expf(m_run[r] - Mn);
      float rs = 0.0f;
      for (int nt = 0; nt < 4; ++nt) {
        const float p = __expf(st[nt][r] - Mn);
        st[nt][r] = p;
        rs += p;
      }
      for (int off = 1; off < 16; off <<= 1) rs += __shfl_xor(rs, off, 32);
      l_run[r] = l_run[r] * alpha + rs;
      m_run[r] = Mn;
      for (int nt = 0; nt < 4; ++nt) o[nt][r] *= alpha;
    }

    // ---- C-layout P -> row-major per-wave LDS (in-order DS per wave) ----
    for (int nt = 0; nt < 4; ++nt)
      for (int r = 0; r < 8; ++r)
        sP[wave][half * 8 + r][nt * 16 + l16] = (bf16)st[nt][r];

    // ---- O(16x64) += P(16x64) @ V_blk(64x64) ----
    for (int f = 0; f < 2; ++f) {
      FragU pa;
      pa.q[0] =
          *reinterpret_cast<const uint4*>(&sP[wave][l16][f * 32 + half * 8]);
      pa.q[1] = *reinterpret_cast<const uint4*>(
          &sP[wave][l16][f * 32 + 16 + half * 8]);
      for (int nt = 0; nt < 4; ++nt) {
        FragU bV;
        bV.q[0] = *reinterpret_cast<const uint4*>(
            &sVt[nt * 16 + l16][f * 32 + half * 8]);
        bV.q[1] = *reinterpret_cast<const uint4*>(
            &sVt[nt * 16 + l16][f * 32 + 16 + half * 8]);
        o[nt] = wmma_bf16(pa.v, bV.v, o[nt]);
      }
    }
    __syncthreads();
  }

  for (int nt = 0; nt < 4; ++nt)
    for (int r = 0; r < 8; ++r) {
      const int qrow = q0 + half * 8 + r;
      const float v  = o[nt][r] / l_run[r];
      ctx[(size_t)(b * S + qrow) * D_MODEL + h * HEAD_DIM + nt * 16 + l16] =
          (bf16)v;
    }
}

// ---------------------------------------------------------------------------
// out = LayerNorm(x + y) * g + b  (one block per row, D_MODEL = 1024)
// ---------------------------------------------------------------------------
__global__ __launch_bounds__(256) void ln_residual(
    const float* __restrict__ x, const float* __restrict__ y,
    const float* __restrict__ g, const float* __restrict__ bt,
    float* __restrict__ of, bf16* __restrict__ ob)
{
  __shared__ float wsum[8], wsq[8];
  const int row  = blockIdx.x;
  const int tid  = threadIdx.x;
  const int lane = tid & 31, wave = tid >> 5;
  float vals[4];
  float s = 0.0f, ss = 0.0f;
  for (int i = 0; i < 4; ++i) {
    const int c   = tid + i * 256;
    const float v = x[(size_t)row * D_MODEL + c] + y[(size_t)row * D_MODEL + c];
    vals[i] = v;
    s += v;
    ss += v * v;
  }
  for (int off = 1; off < 32; off <<= 1) {
    s  += __shfl_xor(s, off, 32);
    ss += __shfl_xor(ss, off, 32);
  }
  if (lane == 0) { wsum[wave] = s; wsq[wave] = ss; }
  __syncthreads();
  if (tid == 0) {
    float S = 0.0f, SS = 0.0f;
    for (int w = 0; w < 8; ++w) { S += wsum[w]; SS += wsq[w]; }
    const float mu  = S / (float)D_MODEL;
    const float var = SS / (float)D_MODEL - mu * mu;
    wsum[0] = mu;
    wsq[0]  = rsqrtf(var + 1e-12f);
  }
  __syncthreads();
  const float mu = wsum[0], inv = wsq[0];
  for (int i = 0; i < 4; ++i) {
    const int c   = tid + i * 256;
    const float v = (vals[i] - mu) * inv * g[c] + bt[c];
    if (of) of[(size_t)row * D_MODEL + c] = v;
    if (ob) ob[(size_t)row * D_MODEL + c] = (bf16)v;
  }
}

// ---------------------------------------------------------------------------
// Host orchestration
// ---------------------------------------------------------------------------
extern "C" void kernel_launch(void* const* d_in, const int* in_sizes, int n_in,
                              void* d_out, int out_size, void* d_ws,
                              size_t ws_size, hipStream_t stream) {
  (void)in_sizes; (void)n_in; (void)out_size; (void)ws_size;

  const float* x    = (const float*)d_in[0];
  const float* mask = (const float*)d_in[1];
  const float* Wq   = (const float*)d_in[2];
  const float* bq   = (const float*)d_in[3];
  const float* Wk   = (const float*)d_in[4];
  const float* bk   = (const float*)d_in[5];
  const float* Wv   = (const float*)d_in[6];
  const float* bv   = (const float*)d_in[7];
  const float* Wo   = (const float*)d_in[8];
  const float* bo   = (const float*)d_in[9];
  const float* W1   = (const float*)d_in[10];
  const float* b1   = (const float*)d_in[11];
  const float* W2   = (const float*)d_in[12];
  const float* b2   = (const float*)d_in[13];
  const float* ln1g = (const float*)d_in[14];
  const float* ln1b = (const float*)d_in[15];
  const float* ln2g = (const float*)d_in[16];
  const float* ln2b = (const float*)d_in[17];
  float* out = (float*)d_out;

  const int B = 2, S = 2048, M = B * S;

  char*  base = (char*)d_ws;
  size_t off  = 0;
  auto alloc = [&](size_t bytes) -> void* {
    void* p = base + off;
    off += bytes;
    off = (off + 255) & ~(size_t)255;
    return p;
  };

  bf16* xb   = (bf16*)alloc((size_t)M * D_MODEL * 2);
  bf16* wqb  = (bf16*)alloc((size_t)D_MODEL * D_MODEL * 2);
  bf16* wkb  = (bf16*)alloc((size_t)D_MODEL * D_MODEL * 2);
  bf16* wvb  = (bf16*)alloc((size_t)D_MODEL * D_MODEL * 2);
  bf16* wob  = (bf16*)alloc((size_t)D_MODEL * D_MODEL * 2);
  bf16* w1b  = (bf16*)alloc((size_t)D_MODEL * D_FF * 2);
  bf16* w2b  = (bf16*)alloc((size_t)D_FF * D_MODEL * 2);
  bf16* qbuf = (bf16*)alloc((size_t)M * D_MODEL * 2);
  bf16* kbuf = (bf16*)alloc((size_t)M * D_MODEL * 2);
  bf16* vbuf = (bf16*)alloc((size_t)M * D_MODEL * 2);
  bf16* ctxb = (bf16*)alloc((size_t)M * D_MODEL * 2);
  float* tmpf = (float*)alloc((size_t)M * D_MODEL * 4);  // attn_out, then ffn_out
  float* x1f  = (float*)alloc((size_t)M * D_MODEL * 4);
  bf16*  x1b  = (bf16*)alloc((size_t)M * D_MODEL * 2);
  bf16*  hb   = (bf16*)alloc((size_t)M * D_FF * 2);

  auto cvt = [&](const float* src, bf16* dst, size_t n) {
    cvt_f32_bf16<<<dim3((unsigned)((n + 255) / 256)), dim3(256), 0, stream>>>(
        src, dst, (int)n);
  };

  // 1) fp32 -> bf16 for activations and weights
  cvt(x, xb, (size_t)M * D_MODEL);
  cvt(Wq, wqb, (size_t)D_MODEL * D_MODEL);
  cvt(Wk, wkb, (size_t)D_MODEL * D_MODEL);
  cvt(Wv, wvb, (size_t)D_MODEL * D_MODEL);
  cvt(Wo, wob, (size_t)D_MODEL * D_MODEL);
  cvt(W1, w1b, (size_t)D_MODEL * D_FF);
  cvt(W2, w2b, (size_t)D_FF * D_MODEL);

  const dim3 blk(256);
  const unsigned g1024 = (unsigned)((M / 128) * (D_MODEL / 64));  // 512
  const unsigned g4096 = (unsigned)((M / 128) * (D_FF / 64));     // 2048

  // 2) QKV projections (bf16 out for attention)
  gemm_bf16_wmma<<<dim3(g1024), blk, 0, stream>>>(xb, wqb, bq, nullptr, qbuf,
                                                  M, D_MODEL, D_MODEL, 0);
  gemm_bf16_wmma<<<dim3(g1024), blk, 0, stream>>>(xb, wkb, bk, nullptr, kbuf,
                                                  M, D_MODEL, D_MODEL, 0);
  gemm_bf16_wmma<<<dim3(g1024), blk, 0, stream>>>(xb, wvb, bv, nullptr, vbuf,
                                                  M, D_MODEL, D_MODEL, 0);

  // 3) flash attention -> ctx (bf16)
  flash_attn_wmma<<<dim3((unsigned)(B * N_HEADS * (S / 128))), blk, 0, stream>>>(
      qbuf, kbuf, vbuf, mask, ctxb, S);

  // 4) output projection (f32 out)
  gemm_bf16_wmma<<<dim3(g1024), blk, 0, stream>>>(ctxb, wob, bo, tmpf, nullptr,
                                                  M, D_MODEL, D_MODEL, 0);

  // 5) residual + LN1 -> x1 (f32 + bf16)
  ln_residual<<<dim3((unsigned)M), blk, 0, stream>>>(x, tmpf, ln1g, ln1b, x1f,
                                                     x1b);

  // 6) FFN up + GELU (bf16 out)
  gemm_bf16_wmma<<<dim3(g4096), blk, 0, stream>>>(x1b, w1b, b1, nullptr, hb,
                                                  M, D_FF, D_MODEL, 1);

  // 7) FFN down (f32 out, reuse tmpf)
  gemm_bf16_wmma<<<dim3(g1024), blk, 0, stream>>>(hb, w2b, b2, tmpf, nullptr,
                                                  M, D_MODEL, D_FF, 0);

  // 8) residual + LN2 -> final output
  ln_residual<<<dim3((unsigned)M), blk, 0, stream>>>(x1f, tmpf, ln2g, ln2b,
                                                     out, nullptr);
}
